// AllSetTransformerLayer_14388140441986
// MI455X (gfx1250) — matrix operand order repaired and checked by reading, not compiled, not run on hardware
//
#include <hip/hip_runtime.h>

// ---------------------------------------------------------------------------
// AllSetTransformer layer for MI455X (gfx1250, wave32, WMMA).
// GEMMs in bf16 WMMA (v_wmma_f32_16x16x32_bf16), f32 accumulate.
// Each wave computes a 16x32 C tile (2 wmma / k-step, shared A fragment).
// Row-clamped tail handling keeps EXEC all-ones through the WMMA loop.
// v2e aggregation exploits contiguous incidence (16 nodes / hyperedge).
// e2v aggregation: 3-pass segment softmax, wave-per-nnz scatter, L2 atomics.
// ---------------------------------------------------------------------------

typedef __bf16 bf16_t;
typedef __bf16 v16bf __attribute__((ext_vector_type(16)));
typedef float  v8f   __attribute__((ext_vector_type(8)));

static constexpr int DNODES = 50000;
static constexpr int DHE    = 25000;
static constexpr int DNNZ   = 400000;
static constexpr int DEG    = 16;
static constexpr int IN0    = 128;   // v2e input channels
static constexpr int DCH    = 256;   // H*C = 4*64

// ------------------------------- utilities ---------------------------------

__global__ void k_cast_bf16(const float* __restrict__ in, bf16_t* __restrict__ out, size_t n) {
  size_t i = (size_t)blockIdx.x * blockDim.x + threadIdx.x;
  size_t st = (size_t)gridDim.x * blockDim.x;
  for (; i < n; i += st) out[i] = (bf16_t)in[i];
}

__global__ void k_zero_u32(unsigned* __restrict__ p, size_t n) {
  size_t i = (size_t)blockIdx.x * blockDim.x + threadIdx.x;
  size_t st = (size_t)gridDim.x * blockDim.x;
  for (; i < n; i += st) p[i] = 0u;
}

// agg[r*256 + c] = att[c]  (skip-connection init for scatter target)
__global__ void k_fill_att(float* __restrict__ agg, const float* __restrict__ att, size_t n) {
  size_t i = (size_t)blockIdx.x * blockDim.x + threadIdx.x;
  size_t st = (size_t)gridDim.x * blockDim.x;
  for (; i < n; i += st) agg[i] = att[i & 255];
}

// V [H=4, In, C=64]  ->  Vt bf16 [256][In]   (row o = h*64+c, K-contiguous)
__global__ void k_vt(const float* __restrict__ V, bf16_t* __restrict__ Vt, int In) {
  int idx = blockIdx.x * blockDim.x + threadIdx.x;
  if (idx >= 256 * In) return;
  int o = idx / In, i = idx % In;
  Vt[idx] = (bf16_t)V[((size_t)(o >> 6) * In + i) * 64 + (o & 63)];
}

// W [256,256] -> Wt bf16 [256][256] transposed (Wt[o][i] = W[i][o])
__global__ void k_wt(const float* __restrict__ W, bf16_t* __restrict__ Wt) {
  int idx = blockIdx.x * blockDim.x + threadIdx.x;
  if (idx >= 256 * 256) return;
  int o = idx >> 8, i = idx & 255;
  Wt[idx] = (bf16_t)W[i * 256 + o];
}

// katt[i][h] = sum_c K[h][i][c] * att[h][0][c]   (folds xK away since Q==1)
__global__ void k_katt(const float* __restrict__ K, const float* __restrict__ att,
                       float* __restrict__ katt, int In) {
  int idx = blockIdx.x * blockDim.x + threadIdx.x;
  if (idx >= In * 4) return;
  int h = idx & 3, i = idx >> 2;
  const float* Kh = K + ((size_t)h * In + i) * 64;
  const float* ah = att + h * 64;
  float s = 0.f;
#pragma unroll
  for (int c = 0; c < 64; c++) s += Kh[c] * ah[c];
  katt[i * 4 + h] = s;
}

// alpha[n][h] = x[n,:] . katt[:,h]
__global__ void k_alpha(const float* __restrict__ x, const float* __restrict__ katt,
                        float* __restrict__ alpha, int rows, int In) {
  int idx = blockIdx.x * blockDim.x + threadIdx.x;
  if (idx >= rows * 4) return;
  int h = idx & 3, n = idx >> 2;
  const float* xr = x + (size_t)n * In;
  float s = 0.f;
  for (int i = 0; i < In; i++) s += xr[i] * katt[i * 4 + h];
  alpha[idx] = s;
}

// ------------------------------ WMMA GEMM ----------------------------------
// C[M,N] = A_bf16[M,K] x Bt_bf16[N][K]  (+bias, optional relu)
// Block = 256 threads = 8 waves as 4 (rows) x 2 (cols); each wave owns a
// 16x32 C tile -> 2 wmma per k-step reusing one A fragment. Block tile 64x64.
// Tail rows are CLAMPED (not branched) so EXEC stays all-ones in the loop;
// stores are guarded instead.
// Fragment layouts used here (A and B share one contiguous k order):
//   A 16x32 bf16: lane<16 -> row=lane, k = 16*half + 0..15 (contig in A row)
//   B 32x16 bf16: lane -> col=lane&15, k = 16*half + 0..15 (contig in Bt row)
//   C 16x16 f32 : vgpr v -> row = v + 8*half, col = lane&15
__global__ __launch_bounds__(256)
void k_gemm_bf16(const bf16_t* __restrict__ A, const bf16_t* __restrict__ Bt,
                 const float* __restrict__ bias,
                 float* __restrict__ outF, bf16_t* __restrict__ outB,
                 int M, int N, int K, int relu)
{
  int w    = threadIdx.x >> 5;
  int lane = threadIdx.x & 31;
  int half = lane >> 4;
  int l16  = lane & 15;
  int rowBase = blockIdx.x * 64 + (w & 3) * 16;
  int colBase = blockIdx.y * 64 + (w >> 2) * 32;

  v8f acc0 = {}, acc1 = {};
  int r  = rowBase + l16;
  int rc = (r < M) ? r : (M - 1);             // clamp -> no exec divergence
  const bf16_t* arow  = A  + (size_t)rc * K;
  const bf16_t* brow0 = Bt + (size_t)(colBase + l16) * K + half * 16;
  const bf16_t* brow1 = Bt + (size_t)(colBase + 16 + l16) * K + half * 16;

  for (int k = 0; k < K; k += 32) {
    union { v16bf v; float4 q[2]; } a, b0, b1;
    a.q[0]  = *reinterpret_cast<const float4*>(arow + k + half * 16);     // k = 16h..16h+7
    a.q[1]  = *reinterpret_cast<const float4*>(arow + k + half * 16 + 8); // k = 16h+8..
    b0.q[0] = *reinterpret_cast<const float4*>(brow0 + k);
    b0.q[1] = *reinterpret_cast<const float4*>(brow0 + k + 8);
    b1.q[0] = *reinterpret_cast<const float4*>(brow1 + k);
    b1.q[1] = *reinterpret_cast<const float4*>(brow1 + k + 8);
    acc0 = __builtin_amdgcn_wmma_f32_16x16x32_bf16(false, a.v, false, b0.v,
                                                   (short)0, acc0, false, false);
    acc1 = __builtin_amdgcn_wmma_f32_16x16x32_bf16(false, a.v, false, b1.v,
                                                   (short)0, acc1, false, false);
  }

  int col0 = colBase + l16;
  int col1 = colBase + 16 + l16;
  float bv0 = bias ? bias[col0] : 0.f;
  float bv1 = bias ? bias[col1] : 0.f;
#pragma unroll
  for (int v = 0; v < 8; v++) {
    int row = rowBase + v + half * 8;
    if (row < M) {
      float x0 = acc0[v] + bv0;
      float x1 = acc1[v] + bv1;
      if (relu) { x0 = fmaxf(x0, 0.f); x1 = fmaxf(x1, 0.f); }
      size_t o0 = (size_t)row * N + col0;
      size_t o1 = (size_t)row * N + col1;
      if (outF) { outF[o0] = x0; outF[o1] = x1; }
      if (outB) { outB[o0] = (bf16_t)x0; outB[o1] = (bf16_t)x1; }
    }
  }
}

// --------------------------- layernorm helpers -----------------------------

__device__ __forceinline__ float block_sum_256(float v, float* red) {
  int t = threadIdx.x;
  red[t] = v; __syncthreads();
#pragma unroll
  for (int s = 128; s > 0; s >>= 1) { if (t < s) red[t] += red[t + s]; __syncthreads(); }
  float r = red[0]; __syncthreads();
  return r;
}

// y -> LN(y)*g+b, write f32 (+ optional bf16)
__global__ __launch_bounds__(256)
void k_ln_row(const float* __restrict__ Y, const float* __restrict__ g,
              const float* __restrict__ b, float* __restrict__ outF,
              bf16_t* __restrict__ outB)
{
  __shared__ float red[256];
  int row = blockIdx.x, t = threadIdx.x;
  float v = Y[(size_t)row * 256 + t];
  float mean = block_sum_256(v, red) * (1.f / 256.f);
  float dv = v - mean;
  float var = block_sum_256(dv * dv, red) * (1.f / 256.f);
  float o = dv * rsqrtf(var + 1e-5f) * g[t] + b[t];
  outF[(size_t)row * 256 + t] = o;
  if (outB) outB[(size_t)row * 256 + t] = (bf16_t)o;
}

// z = yln + relu(ffo); out = relu(LN(z)*g+b)   (conv epilogue, writes d_out)
__global__ __launch_bounds__(256)
void k_ln_resid(const float* __restrict__ Yln, const float* __restrict__ Ffo,
                const float* __restrict__ g, const float* __restrict__ b,
                float* __restrict__ outF, bf16_t* __restrict__ outB)
{
  __shared__ float red[256];
  int row = blockIdx.x, t = threadIdx.x;
  size_t o = (size_t)row * 256 + t;
  float v = Yln[o] + fmaxf(Ffo[o], 0.f);
  float mean = block_sum_256(v, red) * (1.f / 256.f);
  float dv = v - mean;
  float var = block_sum_256(dv * dv, red) * (1.f / 256.f);
  float z = fmaxf(dv * rsqrtf(var + 1e-5f) * g[t] + b[t], 0.f);
  outF[o] = z;
  if (outB) outB[o] = (bf16_t)z;
}

// ------------------- v2e: fused gather/softmax/agg/LN ----------------------
// One block per hyperedge; its 16 sources are contiguous in node_idx.
__global__ __launch_bounds__(256)
void k_v2e_agg(const float* __restrict__ alpha, const float* __restrict__ xV,
               const int* __restrict__ node_idx, const float* __restrict__ att,
               const float* __restrict__ g, const float* __restrict__ b,
               float* __restrict__ yln, bf16_t* __restrict__ ylnb)
{
  __shared__ float w[DEG][4];
  __shared__ int   srcs[DEG];
  __shared__ float red[256];
  int he = blockIdx.x, t = threadIdx.x;
  if (t < DEG) srcs[t] = node_idx[he * DEG + t];
  __syncthreads();
  if (t < 4) {                     // per-head softmax over the 16 members
    float e[DEG], mx = -1e30f;
#pragma unroll
    for (int j = 0; j < DEG; j++) { e[j] = alpha[srcs[j] * 4 + t]; mx = fmaxf(mx, e[j]); }
    float den = 0.f;
#pragma unroll
    for (int j = 0; j < DEG; j++) { e[j] = __expf(e[j] - mx); den += e[j]; }
    float inv = 1.f / den;
#pragma unroll
    for (int j = 0; j < DEG; j++) w[j][t] = e[j] * inv;
  }
  __syncthreads();
  int h = t >> 6;
  float acc = att[t];              // +att skip connection
#pragma unroll
  for (int j = 0; j < DEG; j++) acc += w[j][h] * xV[(size_t)srcs[j] * 256 + t];
  // fused LayerNorm0
  float mean = block_sum_256(acc, red) * (1.f / 256.f);
  float dv = acc - mean;
  float var = block_sum_256(dv * dv, red) * (1.f / 256.f);
  float o = dv * rsqrtf(var + 1e-5f) * g[t] + b[t];
  yln[(size_t)he * 256 + t] = o;
  ylnb[(size_t)he * 256 + t] = (bf16_t)o;
}

// ------------------- e2v: segment softmax with atomics ---------------------

__device__ __forceinline__ unsigned f2key(float f) {
  unsigned bi = __float_as_uint(f);
  return bi ^ ((bi >> 31) ? 0xFFFFFFFFu : 0x80000000u);
}
__device__ __forceinline__ float key2f(unsigned k) {
  unsigned bi = (k >> 31) ? (k ^ 0x80000000u) : ~k;
  return __uint_as_float(bi);
}

__global__ void k_seg_max(const float* __restrict__ alpha, const int* __restrict__ src,
                          const int* __restrict__ tgt, unsigned* __restrict__ segmax, int nnz) {
  int idx = blockIdx.x * blockDim.x + threadIdx.x;
  if (idx >= nnz * 4) return;
  int h = idx & 3, e = idx >> 2;
  atomicMax(&segmax[tgt[e] * 4 + h], f2key(alpha[src[e] * 4 + h]));
}

__global__ void k_seg_den(const float* __restrict__ alpha, const int* __restrict__ src,
                          const int* __restrict__ tgt, const unsigned* __restrict__ segmax,
                          float* __restrict__ den, int nnz) {
  int idx = blockIdx.x * blockDim.x + threadIdx.x;
  if (idx >= nnz * 4) return;
  int h = idx & 3, e = idx >> 2;
  int d = tgt[e] * 4 + h;
  atomicAdd(&den[d], __expf(alpha[src[e] * 4 + h] - key2f(segmax[d])));
}

// agg[tgt] += w * xV[src].  One wave32 per nnz: lane covers 8 channels of one
// head (h = lane>>3), weight recomputed per-lane (cache-broadcast), no LDS,
// no barriers. Accumulator (51 MB) is L2-resident on the 192 MB L2.
__global__ __launch_bounds__(256)
void k_scatter(const float* __restrict__ alpha, const unsigned* __restrict__ segmax,
               const float* __restrict__ den, const float* __restrict__ xV,
               const int* __restrict__ src, const int* __restrict__ tgt,
               float* __restrict__ agg, int nnz)
{
  int wv = (int)((blockIdx.x * blockDim.x + threadIdx.x) >> 5);
  if (wv >= nnz) return;                       // uniform per-wave exit
  int lane = threadIdx.x & 31;
  int s = src[wv], d = tgt[wv];
  int h = lane >> 3;                           // head owning this 8-ch slice
  int di = d * 4 + h;
  float wgt = __expf(alpha[s * 4 + h] - key2f(segmax[di])) / den[di];
  const float* xr = xV + (size_t)s * 256 + lane * 8;
  float* ar = agg + (size_t)d * 256 + lane * 8;
#pragma unroll
  for (int j = 0; j < 8; j++) atomicAdd(&ar[j], wgt * xr[j]);
}

// ------------------------------- launcher ----------------------------------

extern "C" void kernel_launch(void* const* d_in, const int* in_sizes, int n_in,
                              void* d_out, int out_size, void* d_ws, size_t ws_size,
                              hipStream_t stream)
{
  (void)in_sizes; (void)n_in; (void)out_size;
  // pytree (sorted-key) flatten: x_0, e2v{K,V,W1,W2,att,b1,b2,ln0_b,ln0_g,ln1_b,ln1_g},
  //                              v2e{...}, node_idx, he_idx, n_nodes, n_he
  const float* x0    = (const float*)d_in[0];
  const float* K1    = (const float*)d_in[1];
  const float* V1    = (const float*)d_in[2];
  const float* W1_1  = (const float*)d_in[3];
  const float* W2_1  = (const float*)d_in[4];
  const float* att1  = (const float*)d_in[5];
  const float* b1_1  = (const float*)d_in[6];
  const float* b2_1  = (const float*)d_in[7];
  const float* ln0b1 = (const float*)d_in[8];
  const float* ln0g1 = (const float*)d_in[9];
  const float* ln1b1 = (const float*)d_in[10];
  const float* ln1g1 = (const float*)d_in[11];
  const float* K0    = (const float*)d_in[12];
  const float* V0    = (const float*)d_in[13];
  const float* W1_0  = (const float*)d_in[14];
  const float* W2_0  = (const float*)d_in[15];
  const float* att0  = (const float*)d_in[16];
  const float* b1_0  = (const float*)d_in[17];
  const float* b2_0  = (const float*)d_in[18];
  const float* ln0b0 = (const float*)d_in[19];
  const float* ln0g0 = (const float*)d_in[20];
  const float* ln1b0 = (const float*)d_in[21];
  const float* ln1g0 = (const float*)d_in[22];
  const int* node_idx = (const int*)d_in[23];
  const int* he_idx   = (const int*)d_in[24];

  // ---- workspace arena (byte offsets, 256B aligned) ----
  char* ws = (char*)d_ws;
  size_t off = 0;
  auto alloc = [&](size_t bytes) { size_t o = off; off += (bytes + 255) & ~(size_t)255; return o; };
  size_t o_vt0   = alloc((size_t)DCH * IN0 * 2);
  size_t o_vt1   = alloc((size_t)DCH * DCH * 2);
  size_t o_w1t0  = alloc((size_t)DCH * DCH * 2);
  size_t o_w2t0  = alloc((size_t)DCH * DCH * 2);
  size_t o_w1t1  = alloc((size_t)DCH * DCH * 2);
  size_t o_w2t1  = alloc((size_t)DCH * DCH * 2);
  size_t o_katt0 = alloc((size_t)IN0 * 4 * 4);
  size_t o_katt1 = alloc((size_t)DCH * 4 * 4);
  size_t o_x0bf  = alloc((size_t)DNODES * IN0 * 2);
  size_t o_alpha = alloc((size_t)DNODES * 4 * 4);        // alpha0, then alpha1
  size_t o_xvagg = alloc((size_t)DNODES * DCH * 4);      // xV0, then e2v agg
  size_t o_ylnA  = alloc((size_t)DHE * DCH * 4);         // yln0, then xV1
  size_t o_ylnbf = alloc((size_t)DNODES * DCH * 2);      // yln bf16 (both convs)
  size_t o_h1bf  = alloc((size_t)DNODES * DCH * 2);      // rFF hidden (both convs)
  size_t o_ffo   = alloc((size_t)DNODES * DCH * 4);      // rFF output (both convs)
  size_t o_x1bf  = alloc((size_t)DHE * DCH * 2);
  size_t o_seg   = alloc((size_t)DNODES * 8 * 4);        // segmax (200k u32) + den (200k f32)
  size_t o_yln1  = alloc((size_t)DNODES * DCH * 4);
  if (off > ws_size) return;                             // scratch insufficient

  bf16_t* vt0   = (bf16_t*)(ws + o_vt0);
  bf16_t* vt1   = (bf16_t*)(ws + o_vt1);
  bf16_t* w1t0  = (bf16_t*)(ws + o_w1t0);
  bf16_t* w2t0  = (bf16_t*)(ws + o_w2t0);
  bf16_t* w1t1  = (bf16_t*)(ws + o_w1t1);
  bf16_t* w2t1  = (bf16_t*)(ws + o_w2t1);
  float*  katt0 = (float*)(ws + o_katt0);
  float*  katt1 = (float*)(ws + o_katt1);
  bf16_t* x0bf  = (bf16_t*)(ws + o_x0bf);
  float*  alpha = (float*)(ws + o_alpha);
  float*  xvagg = (float*)(ws + o_xvagg);
  float*  ylnA  = (float*)(ws + o_ylnA);
  bf16_t* ylnbf = (bf16_t*)(ws + o_ylnbf);
  bf16_t* h1bf  = (bf16_t*)(ws + o_h1bf);
  float*  ffo   = (float*)(ws + o_ffo);
  bf16_t* x1bf  = (bf16_t*)(ws + o_x1bf);
  unsigned* segmax = (unsigned*)(ws + o_seg);
  float*  den   = (float*)(ws + o_seg) + (size_t)DNODES * 4;
  float*  yln1  = (float*)(ws + o_yln1);

  float* x0out = (float*)d_out;                          // [50000,256]
  float* x1    = (float*)d_out + (size_t)DNODES * DCH;   // [25000,256]

  dim3 B(256);
  // ---- phase 0: weight prep ----
  k_vt  <<<(DCH * IN0 + 255) / 256, B, 0, stream>>>(V0, vt0, IN0);
  k_vt  <<<(DCH * DCH + 255) / 256, B, 0, stream>>>(V1, vt1, DCH);
  k_wt  <<<DCH * DCH / 256, B, 0, stream>>>(W1_0, w1t0);
  k_wt  <<<DCH * DCH / 256, B, 0, stream>>>(W2_0, w2t0);
  k_wt  <<<DCH * DCH / 256, B, 0, stream>>>(W1_1, w1t1);
  k_wt  <<<DCH * DCH / 256, B, 0, stream>>>(W2_1, w2t1);
  k_katt<<<(IN0 * 4 + 255) / 256, B, 0, stream>>>(K0, att0, katt0, IN0);
  k_katt<<<(DCH * 4 + 255) / 256, B, 0, stream>>>(K1, att1, katt1, DCH);
  k_cast_bf16<<<1024, B, 0, stream>>>(x0, x0bf, (size_t)DNODES * IN0);

  // ---- conv 1 : v2e (targets = hyperedges) ----
  k_alpha<<<(DNODES * 4 + 255) / 256, B, 0, stream>>>(x0, katt0, alpha, DNODES, IN0);
  k_gemm_bf16<<<dim3((DNODES + 63) / 64, DCH / 64), B, 0, stream>>>(
      x0bf, vt0, nullptr, xvagg, nullptr, DNODES, DCH, IN0, 0);
  k_v2e_agg<<<DHE, B, 0, stream>>>(alpha, xvagg, node_idx, att0, ln0g0, ln0b0, ylnA, ylnbf);
  k_gemm_bf16<<<dim3((DHE + 63) / 64, DCH / 64), B, 0, stream>>>(
      ylnbf, w1t0, b1_0, nullptr, h1bf, DHE, DCH, DCH, 1);
  k_gemm_bf16<<<dim3((DHE + 63) / 64, DCH / 64), B, 0, stream>>>(
      h1bf, w2t0, b2_0, ffo, nullptr, DHE, DCH, DCH, 0);
  k_ln_resid<<<DHE, B, 0, stream>>>(ylnA, ffo, ln1g0, ln1b0, x1, x1bf);

  // ---- conv 2 : e2v (targets = nodes) ----
  k_alpha<<<(DHE * 4 + 255) / 256, B, 0, stream>>>(x1, katt1, alpha, DHE, DCH);
  k_gemm_bf16<<<dim3((DHE + 63) / 64, DCH / 64), B, 0, stream>>>(
      x1bf, vt1, nullptr, ylnA /*xV1*/, nullptr, DHE, DCH, DCH, 0);
  k_fill_att<<<2048, B, 0, stream>>>(xvagg /*agg*/, att1, (size_t)DNODES * DCH);
  k_zero_u32<<<(DNODES * 8 + 255) / 256, B, 0, stream>>>(segmax, (size_t)DNODES * 8);
  k_seg_max<<<(DNNZ * 4 + 255) / 256, B, 0, stream>>>(alpha, he_idx, node_idx, segmax, DNNZ);
  k_seg_den<<<(DNNZ * 4 + 255) / 256, B, 0, stream>>>(alpha, he_idx, node_idx, segmax, den, DNNZ);
  k_scatter<<<(DNNZ * 32 + 255) / 256, B, 0, stream>>>(
      alpha, segmax, den, ylnA /*xV1*/, he_idx, node_idx, xvagg, DNNZ);
  k_ln_row<<<DNODES, B, 0, stream>>>(xvagg, ln0g1, ln0b1, yln1, ylnbf);
  k_gemm_bf16<<<dim3((DNODES + 63) / 64, DCH / 64), B, 0, stream>>>(
      ylnbf, w1t1, b1_1, nullptr, h1bf, DNODES, DCH, DCH, 1);
  k_gemm_bf16<<<dim3((DNODES + 63) / 64, DCH / 64), B, 0, stream>>>(
      h1bf, w2t1, b2_1, ffo, nullptr, DNODES, DCH, DCH, 0);
  k_ln_resid<<<DNODES, B, 0, stream>>>(yln1, ffo, ln1g1, ln1b1, x0out, nullptr);
}
